// run_complex_network_85813446574648
// MI455X (gfx1250) — matrix operand order repaired and checked by reading, not compiled, not run on hardware
//
#include <hip/hip_runtime.h>

// ===================== problem constants =====================
constexpr int S_ = 256;          // sequence length
constexpr int B_ = 8;            // batch
constexpr int E_ = 256;          // embedding
constexpr int G_ = 4 * E_;       // 1024 = gate width (i,f,g,o)
constexpr int OUT_ = 2;
constexpr long SBE  = (long)S_ * B_ * E_;   // 524288
constexpr long PREN = (long)S_ * B_ * G_;   // 2097152
constexpr long EE   = (long)E_ * E_;        // 65536
constexpr long SEl  = (long)S_ * E_;        // 65536
constexpr int  BS   = B_ * S_;              // 2048

typedef __attribute__((ext_vector_type(16))) _Float16 v16h;
typedef __attribute__((ext_vector_type(8)))  _Float16 v8h;
typedef __attribute__((ext_vector_type(8)))  float    v8f;
typedef __attribute__((ext_vector_type(4)))  float    v4f;

__device__ __forceinline__ float sigf(float x) { return 1.0f / (1.0f + __expf(-x)); }

// ===================== generic WMMA GEMM =====================
// C[M,N] = alpha * op(A)[M,K] @ op(B)[K,N] + beta*C (+ bias[n])
// One wave per 16x16 C tile; K stepped by 32 through v_wmma_f32_16x16x32_f16.
// Fragment layouts used here (contiguous K, A and B alike):
//   A 16x32: lane<16 -> M=lane, K=0..15;            lane>=16 -> M=lane-16, K=16..31
//   B 32x16: lane<16 -> N=lane, K=0..15;            lane>=16 -> N=lane-16, K=16..31
// Contiguous directions (row-major A, W^T-style B) use 128-bit vector loads.
template <bool TA, bool TB>
__global__ __launch_bounds__(32)
void k_gemm(const float* __restrict__ A, const float* __restrict__ Bm,
            const float* __restrict__ bias, float* __restrict__ C,
            int lda, int ldb, int ldc, int K,
            long sA, long sB, long sC, float alpha, float beta)
{
    const int lane = threadIdx.x & 31;
    const int hl   = lane >> 4;       // half-wave select
    const int r    = lane & 15;
    const long zb  = blockIdx.z;
    const float* Ab = A  + zb * sA;
    const float* Bb = Bm + zb * sB;
    float*       Cb = C  + zb * sC;
    const int m0 = blockIdx.y * 16;
    const int n0 = blockIdx.x * 16;

    v8f acc = {};
    for (int k0 = 0; k0 < K; k0 += 32) {
        if (k0 + 32 < K) {  // L2-resident streaming hint -> global_prefetch_b8
            __builtin_prefetch(TA ? &Ab[(long)(k0 + 32) * lda + (m0 + r)]
                                  : &Ab[(long)(m0 + r) * lda + (k0 + 32)]);
            __builtin_prefetch(TB ? &Bb[(long)(n0 + r) * ldb + (k0 + 32)]
                                  : &Bb[(long)(k0 + 32) * ldb + (n0 + r)]);
        }
        v16h af, bf;
        if (!TA) {
            // 16 contiguous floats starting at k0+hl*16 -> 4x b128
            const v4f* ar = (const v4f*)(Ab + (long)(m0 + r) * lda + k0);
            const v4f a0 = ar[hl * 4];
            const v4f a1 = ar[hl * 4 + 1];
            const v4f a2 = ar[hl * 4 + 2];
            const v4f a3 = ar[hl * 4 + 3];
#pragma unroll
            for (int e = 0; e < 4; ++e) {
                af[e]      = (_Float16)a0[e];
                af[4 + e]  = (_Float16)a1[e];
                af[8 + e]  = (_Float16)a2[e];
                af[12 + e] = (_Float16)a3[e];
            }
        } else {
#pragma unroll
            for (int e = 0; e < 16; ++e) {
                const int ka = k0 + e + hl * 16;
                af[e] = (_Float16)Ab[(long)ka * lda + (m0 + r)];
            }
        }
        if (TB) {
            // 16 contiguous floats starting at k0+hl*16 -> 4x b128
            const v4f* br = (const v4f*)(Bb + (long)(n0 + r) * ldb + k0 + hl * 16);
            const v4f b0 = br[0];
            const v4f b1 = br[1];
            const v4f b2 = br[2];
            const v4f b3 = br[3];
#pragma unroll
            for (int e = 0; e < 4; ++e) {
                bf[e]      = (_Float16)b0[e];
                bf[4 + e]  = (_Float16)b1[e];
                bf[8 + e]  = (_Float16)b2[e];
                bf[12 + e] = (_Float16)b3[e];
            }
        } else {
#pragma unroll
            for (int e = 0; e < 16; ++e) {
                const int kb = k0 + e + hl * 16;
                bf[e] = (_Float16)Bb[(long)kb * ldb + (n0 + r)];
            }
        }
        acc = __builtin_amdgcn_wmma_f32_16x16x32_f16(false, af, false, bf,
                                                     (short)0, acc, false, false);
    }
    const int col = n0 + r;
#pragma unroll
    for (int i = 0; i < 8; ++i) {
        const int row = m0 + i + hl * 8;        // C/D layout: lanes>=16 hold M=8..15
        float vv = alpha * acc[i];
        if (bias) vv += bias[col];
        if (beta != 0.0f) vv += beta * Cb[(long)row * ldc + col];
        Cb[(long)row * ldc + col] = vv;
    }
}

// ===================== f32 -> f16 weight shadow =====================
__global__ void k_f32_to_f16(const float* __restrict__ x, _Float16* __restrict__ y, long n)
{
    long i = (long)blockIdx.x * blockDim.x + threadIdx.x;
    if (i < n) y[i] = (_Float16)x[i];
}

// ===================== LSTM recurrence (persistent, 1 WGP) =====================
// pre[t,b,n] = (x @ W_ih^T + b_ih + b_hh) precomputed by k_gemm.
// 256 dependent steps; h (f16, padded to 16 rows), c (f32), gates live in LDS.
// 32 waves x 32 gate columns; h @ W_hh^T via WMMA: A fragment = 2x ds_load_b128
// from the f16 hidden state, B fragment = one packed 32B load of the f16
// W_hh shadow (pure f16 inner loop: no converts on the critical path).
__global__ __launch_bounds__(1024)
void k_lstm_scan(const float* __restrict__ pre, const _Float16* __restrict__ w_hh,
                 float* __restrict__ hs)
{
    __shared__ _Float16 h_lds[16][E_];   // rows 8..15 stay zero (WMMA M-padding)
    __shared__ float    c_lds[B_][E_];
    __shared__ float    g_lds[B_][G_];

    const int tid = threadIdx.x;
    for (int i = tid; i < 16 * E_; i += 1024) h_lds[i >> 8][i & 255] = (_Float16)0.0f;
    for (int i = tid; i < B_ * E_; i += 1024) c_lds[i >> 8][i & 255] = 0.0f;
    __syncthreads();

    const int wave = tid >> 5;
    const int lane = tid & 31;
    const int hl   = lane >> 4;
    const int r    = lane & 15;

    for (int t = 0; t < S_; ++t) {
        const float* pre_t = pre + (long)t * (B_ * G_);
#pragma unroll
        for (int tt = 0; tt < 2; ++tt) {
            const int n0 = (wave * 2 + tt) * 16;
            v8f acc = {};
            const v8h* hrow = (const v8h*)(&h_lds[r][0]);
            for (int k0 = 0; k0 < E_; k0 += 32) {
                union { v16h v; v8h h[2]; } ua;
                ua.h[0] = hrow[(k0 >> 3) + hl * 2];      // K = k0+hl*16 .. +8
                ua.h[1] = hrow[(k0 >> 3) + hl * 2 + 1];  // K = k0+hl*16+8 .. +8
                const v16h bf = *(const v16h*)(w_hh + (long)(n0 + r) * E_ + k0 + hl * 16);
                acc = __builtin_amdgcn_wmma_f32_16x16x32_f16(false, ua.v, false, bf,
                                                             (short)0, acc, false, false);
            }
            if (hl == 0) {                 // only M=0..7 (real batch rows) matter
#pragma unroll
                for (int i = 0; i < 8; ++i)
                    g_lds[i][n0 + r] = acc[i] + pre_t[i * G_ + n0 + r];
            }
        }
        __syncthreads();
        for (int j = tid; j < B_ * E_; j += 1024) {
            const int b = j >> 8, e = j & 255;
            const float gi = g_lds[b][e];
            const float gf = g_lds[b][E_ + e];
            const float gg = g_lds[b][2 * E_ + e];
            const float go = g_lds[b][3 * E_ + e];
            const float c  = sigf(gf) * c_lds[b][e] + sigf(gi) * tanhf(gg);
            const float h  = sigf(go) * tanhf(c);
            c_lds[b][e] = c;
            h_lds[b][e] = (_Float16)h;
            hs[(long)t * (B_ * E_) + j] = h;
        }
        __syncthreads();
    }
}

// ===================== small elementwise / reduction kernels =====================
__global__ void k_embed(const int* __restrict__ text, const float* __restrict__ sent,
                        const float* __restrict__ emb, float* __restrict__ amp,
                        float* __restrict__ phase)
{
    long i = (long)blockIdx.x * blockDim.x + threadIdx.x;
    if (i >= SBE) return;
    const long sb = i >> 8;
    const int  e  = (int)(i & 255);
    const int tok = text[sb];
    amp[i]   = emb[(long)tok * E_ + e];
    phase[i] = sent[sb];
}

__global__ void k_transpose_sb(const float* __restrict__ x, float* __restrict__ y)
{   // y[b,s,e] = x[s,b,e]
    long i = (long)blockIdx.x * blockDim.x + threadIdx.x;
    if (i >= SBE) return;
    const int  e  = (int)(i & 255);
    const long sb = i >> 8;
    const int  b  = (int)(sb & 7);
    const long s  = sb >> 3;
    y[((long)b * S_ + s) * E_ + e] = x[i];
}

__global__ void k_vadd(const float* __restrict__ a, const float* __restrict__ b,
                       float* __restrict__ c, int n)
{
    int i = blockIdx.x * blockDim.x + threadIdx.x;
    if (i < n) c[i] = a[i] + b[i];
}

__global__ __launch_bounds__(256)
void k_rownorm(float* __restrict__ x, float eps)
{   // in-place L2-normalize rows of length E_
    __shared__ float red[256];
    const long row = blockIdx.x;
    const int  t   = threadIdx.x;
    const float v  = x[row * E_ + t];
    red[t] = v * v; __syncthreads();
    for (int o = 128; o > 0; o >>= 1) { if (t < o) red[t] += red[t + o]; __syncthreads(); }
    x[row * E_ + t] = v * (1.0f / fmaxf(sqrtf(red[0]), eps));
}

__global__ __launch_bounds__(256)
void k_softmax(float* __restrict__ x, float mul)
{   // softmax over rows of length 256, with pre-scale `mul`
    __shared__ float red[256];
    const long row = blockIdx.x;
    const int  t   = threadIdx.x;
    const float v  = x[row * 256 + t] * mul;
    red[t] = v; __syncthreads();
    for (int o = 128; o > 0; o >>= 1) { if (t < o) red[t] = fmaxf(red[t], red[t + o]); __syncthreads(); }
    const float mx = red[0]; __syncthreads();
    const float ev = __expf(v - mx);
    red[t] = ev; __syncthreads();
    for (int o = 128; o > 0; o >>= 1) { if (t < o) red[t] += red[t + o]; __syncthreads(); }
    x[row * 256 + t] = ev / red[0];
}

__global__ __launch_bounds__(256)
void k_euler(const float* __restrict__ amp2, const float* __restrict__ ph,
             float* __restrict__ re, float* __restrict__ im)
{
    __shared__ float red[256];
    const long row = blockIdx.x;          // b*S + s
    const int  t   = threadIdx.x;
    const long idx = row * E_ + t;
    const float v  = amp2[idx];
    red[t] = v * v; __syncthreads();
    for (int o = 128; o > 0; o >>= 1) { if (t < o) red[t] += red[t + o]; __syncthreads(); }
    const float an = v / fmaxf(sqrtf(red[0]), 1e-12f);
    const float p  = ph[idx];
    re[idx] = an * __cosf(p);
    im[idx] = an * __sinf(p);
}

__global__ __launch_bounds__(256)
void k_logits(const float* __restrict__ r1, const float* __restrict__ i1,
              const float* __restrict__ r2, const float* __restrict__ i2,
              float* __restrict__ logits)   // [0,BS): real ; [BS,2BS): imag
{
    __shared__ float redr[256];
    __shared__ float redi[256];
    const long row = blockIdx.x;
    const int  t   = threadIdx.x;
    const long idx = row * E_ + t;
    const float a = r1[idx], b = i1[idx], c = r2[idx], d = i2[idx];
    const float dr1 = a * a - b * b, di1 = 2.0f * a * b;
    const float dr2 = c * c - d * d, di2 = 2.0f * c * d;
    redr[t] = dr1 * dr2; redi[t] = di1 * di2; __syncthreads();
    for (int o = 128; o > 0; o >>= 1) {
        if (t < o) { redr[t] += redr[t + o]; redi[t] += redi[t + o]; }
        __syncthreads();
    }
    if (t == 0) { logits[row] = redr[0]; logits[BS + row] = redi[0]; }
}

__global__ void k_scale4(const float* __restrict__ a /*2*BS*/,
                         const float* __restrict__ r2, const float* __restrict__ i2,
                         float* __restrict__ wr2, float* __restrict__ wi2,
                         float* __restrict__ ur2, float* __restrict__ ui2)
{
    long i = (long)blockIdx.x * blockDim.x + threadIdx.x;
    if (i >= SBE) return;
    const long row = i >> 8;              // b*S + s
    const float ar = a[row], ai = a[BS + row];
    const float r = r2[i], m = i2[i];
    wr2[i] = ar * r; wi2[i] = ar * m;
    ur2[i] = ai * r; ui2[i] = ai * m;
}

__global__ __launch_bounds__(256)
void k_projnorm(const float* __restrict__ proj, float* __restrict__ prn,
                float* __restrict__ pin)
{   // grid.x = 2 (real / imag column of projector [2,E,1])
    __shared__ float red[256];
    const int t = threadIdx.x;
    const float* p = proj + (long)blockIdx.x * E_;
    const float v = p[t];
    red[t] = v * v; __syncthreads();
    for (int o = 128; o > 0; o >>= 1) { if (t < o) red[t] += red[t + o]; __syncthreads(); }
    const float s = 1.0f / sqrtf(red[0]);
    (blockIdx.x == 0 ? prn : pin)[t] = v * s;
}

__global__ void k_matvec(const float* __restrict__ MR, const float* __restrict__ MI,
                         const float* __restrict__ prn, const float* __restrict__ pin,
                         float* __restrict__ t1, float* __restrict__ t2,
                         float* __restrict__ t3, float* __restrict__ t4)
{   // rank-1 projector contraction: t1=MR·p, t2=MI·q, t3=MR·q, t4=MI·p
    const int i = blockIdx.x * blockDim.x + threadIdx.x;  // b*E + row
    if (i >= B_ * E_) return;
    const float* mr = MR + (long)i * E_;
    const float* mi = MI + (long)i * E_;
    float s1 = 0, s2 = 0, s3 = 0, s4 = 0;
    for (int j = 0; j < E_; ++j) {
        const float p = prn[j], q = pin[j], x = mr[j], y = mi[j];
        s1 += x * p; s3 += x * q; s4 += y * p; s2 += y * q;
    }
    t1[i] = s1; t2[i] = s2; t3[i] = s3; t4[i] = s4;
}

__global__ void k_pv(const float* __restrict__ t1, const float* __restrict__ t2,
                     const float* __restrict__ t3, const float* __restrict__ t4,
                     const float* __restrict__ prn, const float* __restrict__ pin,
                     float* __restrict__ PvR, float* __restrict__ PvI)
{
    long i = (long)blockIdx.x * blockDim.x + threadIdx.x;
    if (i >= (long)B_ * E_ * E_) return;
    const int  j  = (int)(i & 255);
    const long bi = i >> 8;               // b*E + row
    PvR[i] = t1[bi] * prn[j] - t2[bi] * pin[j];
    PvI[i] = t3[bi] * pin[j] + t4[bi] * prn[j];
}

__global__ __launch_bounds__(256)
void k_convpool(const float* __restrict__ PvR, const float* __restrict__ PvI,
                const float* __restrict__ w1, const float* __restrict__ b1,
                const float* __restrict__ w2, const float* __restrict__ b2,
                float* __restrict__ pooled)
{   // grid = (E-2, B, 2); sigmoid(3x3 conv) then max over H -> pooled[b, ch*254 + col]
    __shared__ float red[256];
    const int col = blockIdx.x;
    const int b   = blockIdx.y;
    const int ch  = blockIdx.z;
    const float* src = (ch == 0 ? PvR : PvI) + (long)b * EE;
    const float* w   = (ch == 0 ? w1 : w2);
    const float  bc  = (ch == 0 ? b1 : b2)[0];
    const int t = threadIdx.x;
    float mx = -INFINITY;
    for (int h = t; h < E_ - 2; h += 256) {
        float acc = bc;
#pragma unroll
        for (int dy = 0; dy < 3; ++dy)
#pragma unroll
            for (int dx = 0; dx < 3; ++dx)
                acc += w[dy * 3 + dx] * src[(long)(h + dy) * E_ + (col + dx)];
        mx = fmaxf(mx, sigf(acc));
    }
    red[t] = mx; __syncthreads();
    for (int o = 128; o > 0; o >>= 1) { if (t < o) red[t] = fmaxf(red[t], red[t + o]); __syncthreads(); }
    if (t == 0) pooled[(long)b * (2 * (E_ - 2)) + ch * (E_ - 2) + col] = red[0];
}

__global__ __launch_bounds__(256)
void k_fc1(const float* __restrict__ pooled, const float* __restrict__ w,
           const float* __restrict__ bias, float* __restrict__ h10)
{
    const int t = threadIdx.x;
    if (t >= B_ * 10) return;
    const int b = t / 10, u = t % 10;
    const int F = 2 * (E_ - 2);           // 508
    float s = bias[u];
    const float* f = pooled + (long)b * F;
    for (int j = 0; j < F; ++j) s += f[j] * w[u * F + j];
    h10[t] = sigf(s);
}

__global__ void k_fc2(const float* __restrict__ h10, const float* __restrict__ w,
                      const float* __restrict__ bias, float* __restrict__ out)
{
    const int t = threadIdx.x;
    if (t >= B_ * OUT_) return;
    const int b = t / OUT_, o = t % OUT_;
    float s = bias[o];
    for (int u = 0; u < 10; ++u) s += h10[b * 10 + u] * w[o * 10 + u];
    out[t] = sigf(s);                     // d_out layout [B,1,1,OUT]
}

// ===================== host side =====================
// Input order: insertion-order flattening of setup_inputs() nested dicts.
enum {
    IN_TEXT = 0, IN_SENT,
    IN_EMB,
    L1_WIH0, L1_WHH0, L1_BIH0, L1_BHH0, L1_WIH1, L1_WHH1, L1_BIH1, L1_BHH1, L1_MIDW, L1_MIDB,
    L2_WIH0, L2_WHH0, L2_BIH0, L2_BHH0, L2_WIH1, L2_WHH1, L2_BIH1, L2_BHH1, L2_MIDW, L2_MIDB,
    IN_WQ, IN_WK, IN_WV,
    IN_WR1, IN_BR1, IN_WI1, IN_BI1, IN_WR2, IN_BR2, IN_WI2, IN_BI2,
    IN_PROJ,
    IN_C1W, IN_C1B, IN_C2W, IN_C2B,
    IN_FC1W, IN_FC1B, IN_FC2W, IN_FC2B
};

static void launch_gemm(hipStream_t st, const float* A, const float* Bm,
                        const float* bias, float* C, int M, int N, int K,
                        int lda, int ldb, int ldc,
                        long sA, long sB, long sC, int batch,
                        float alpha, float beta, bool TA, bool TB)
{
    dim3 grid(N / 16, M / 16, batch);
    dim3 blk(32);
    if (!TA && !TB)
        k_gemm<false, false><<<grid, blk, 0, st>>>(A, Bm, bias, C, lda, ldb, ldc, K, sA, sB, sC, alpha, beta);
    else if (!TA && TB)
        k_gemm<false, true><<<grid, blk, 0, st>>>(A, Bm, bias, C, lda, ldb, ldc, K, sA, sB, sC, alpha, beta);
    else if (TA && !TB)
        k_gemm<true, false><<<grid, blk, 0, st>>>(A, Bm, bias, C, lda, ldb, ldc, K, sA, sB, sC, alpha, beta);
    else
        k_gemm<true, true><<<grid, blk, 0, st>>>(A, Bm, bias, C, lda, ldb, ldc, K, sA, sB, sC, alpha, beta);
}

extern "C" void kernel_launch(void* const* d_in, const int* in_sizes, int n_in,
                              void* d_out, int out_size, void* d_ws, size_t ws_size,
                              hipStream_t stream)
{
    (void)in_sizes; (void)n_in; (void)out_size; (void)ws_size;
    auto F = [&](int i) { return (const float*)d_in[i]; };
    const int* text = (const int*)d_in[IN_TEXT];
    const float* sent = F(IN_SENT);

    // ---- workspace layout (floats); total ~11.2M floats (~45 MB) ----
    float* W = (float*)d_ws;
    long off = 0;
    auto alloc = [&](long n) { float* p = W + off; off += n; return p; };
    float* amp   = alloc(SBE);
    float* phase = alloc(SBE);
    float* pre   = alloc(PREN);
    float* h1    = alloc(SBE);
    float* lout  = alloc(SBE);
    float* xb    = alloc(SBE);
    float* ph    = alloc(SBE);
    float* q     = alloc(SBE);
    float* kk    = alloc(SBE);
    float* vv    = alloc(SBE);
    float* attn  = alloc(SBE);   // B*S*S == SBE
    float* amp2  = alloc(SBE);
    float* re    = alloc(SBE);
    float* im    = alloc(SBE);
    float* r1    = alloc(SBE);
    float* i1    = alloc(SBE);
    float* r2    = alloc(SBE);
    float* i2    = alloc(SBE);
    _Float16* whh_h = (_Float16*)alloc((long)G_ * E_ / 2);   // f16 W_hh shadow (32B-aligned)
    float* bias_sum = alloc(G_);
    float* logits   = alloc(2 * BS);
    float* prn      = alloc(E_);
    float* pin      = alloc(E_);
    float* t1 = alloc(B_ * E_);
    float* t2 = alloc(B_ * E_);
    float* t3 = alloc(B_ * E_);
    float* t4 = alloc(B_ * E_);
    float* pooled = alloc(B_ * 2 * (E_ - 2));
    float* h10    = alloc(B_ * 10);
    // dead-buffer aliases (stream-ordered, safe):
    float* wr2 = q;   float* wi2 = kk;   float* ur2 = vv;  float* ui2 = attn;
    float* MR  = amp; float* MI  = phase;
    float* PvR = re;  float* PvI = im;

    const int TPB = 256;
    const long nSBE = SBE;
    const long nBEE = (long)B_ * E_ * E_;
    const long nWHH = (long)G_ * E_;

    // ---- embedding + phase broadcast ----
    k_embed<<<(nSBE + TPB - 1) / TPB, TPB, 0, stream>>>(text, sent, F(IN_EMB), amp, phase);

    // ---- lstm_skip helper ----
    auto lstm_skip = [&](const float* x, int wih0, int whh0, int bih0, int bhh0,
                         int wih1, int whh1, int bih1, int bhh1, int midw, int midb,
                         float* out) {
        k_vadd<<<(G_ + TPB - 1) / TPB, TPB, 0, stream>>>(F(bih0), F(bhh0), bias_sum, G_);
        launch_gemm(stream, x, F(wih0), bias_sum, pre, S_ * B_, G_, E_,
                    E_, E_, G_, 0, 0, 0, 1, 1.0f, 0.0f, false, true);
        k_f32_to_f16<<<(nWHH + TPB - 1) / TPB, TPB, 0, stream>>>(F(whh0), whh_h, nWHH);
        k_lstm_scan<<<1, 1024, 0, stream>>>(pre, whh_h, h1);
        k_vadd<<<(G_ + TPB - 1) / TPB, TPB, 0, stream>>>(F(bih1), F(bhh1), bias_sum, G_);
        launch_gemm(stream, h1, F(wih1), bias_sum, pre, S_ * B_, G_, E_,
                    E_, E_, G_, 0, 0, 0, 1, 1.0f, 0.0f, false, true);
        k_f32_to_f16<<<(nWHH + TPB - 1) / TPB, TPB, 0, stream>>>(F(whh1), whh_h, nWHH);
        k_lstm_scan<<<1, 1024, 0, stream>>>(pre, whh_h, out);
        launch_gemm(stream, x, F(midw), F(midb), out, S_ * B_, E_, E_,
                    E_, E_, E_, 0, 0, 0, 1, 1.0f, 1.0f, false, true);   // skip: out += x@mid^T+b
    };

    // ---- LSTM stack 1 on amplitudes ----
    lstm_skip(amp, L1_WIH0, L1_WHH0, L1_BIH0, L1_BHH0,
              L1_WIH1, L1_WHH1, L1_BIH1, L1_BHH1, L1_MIDW, L1_MIDB, lout);
    k_transpose_sb<<<(nSBE + TPB - 1) / TPB, TPB, 0, stream>>>(lout, xb);

    // ---- attention (batched over B in grid.z) ----
    launch_gemm(stream, xb, F(IN_WQ), nullptr, q,  S_, E_, E_, E_, E_, E_, SEl, 0, SEl, B_, 1.0f, 0.0f, false, false);
    launch_gemm(stream, xb, F(IN_WK), nullptr, kk, S_, E_, E_, E_, E_, E_, SEl, 0, SEl, B_, 1.0f, 0.0f, false, false);
    launch_gemm(stream, xb, F(IN_WV), nullptr, vv, S_, E_, E_, E_, E_, E_, SEl, 0, SEl, B_, 1.0f, 0.0f, false, false);
    k_rownorm<<<BS, 256, 0, stream>>>(q,  1e-20f);
    k_rownorm<<<BS, 256, 0, stream>>>(kk, 1e-20f);
    k_rownorm<<<BS, 256, 0, stream>>>(vv, 1e-20f);
    launch_gemm(stream, q, kk, nullptr, attn, S_, S_, E_, E_, E_, S_,
                SEl, SEl, (long)S_ * S_, B_, 1.0f, 0.0f, false, true);
    k_softmax<<<BS, 256, 0, stream>>>(attn, 0.35355339059327373f);      // 1/sqrt(B)
    launch_gemm(stream, attn, vv, nullptr, amp2, S_, E_, S_, S_, E_, E_,
                (long)S_ * S_, SEl, SEl, B_, 1.0f, 0.0f, false, false);

    // ---- LSTM stack 2 on phases ----
    lstm_skip(phase, L2_WIH0, L2_WHH0, L2_BIH0, L2_BHH0,
              L2_WIH1, L2_WHH1, L2_BIH1, L2_BHH1, L2_MIDW, L2_MIDB, lout);
    k_transpose_sb<<<(nSBE + TPB - 1) / TPB, TPB, 0, stream>>>(lout, ph);

    // ---- Euler projection ----
    k_euler<<<BS, 256, 0, stream>>>(amp2, ph, re, im);

    // ---- complex linears (single M=2048 GEMMs) ----
    launch_gemm(stream, re, F(IN_WR1), F(IN_BR1), r1, BS, E_, E_, E_, E_, E_, 0, 0, 0, 1, 1.0f, 0.0f, false, true);
    launch_gemm(stream, im, F(IN_WI1), F(IN_BI1), i1, BS, E_, E_, E_, E_, E_, 0, 0, 0, 1, 1.0f, 0.0f, false, true);
    launch_gemm(stream, re, F(IN_WR2), F(IN_BR2), r2, BS, E_, E_, E_, E_, E_, 0, 0, 0, 1, 1.0f, 0.0f, false, true);
    launch_gemm(stream, im, F(IN_WI2), F(IN_BI2), i2, BS, E_, E_, E_, E_, E_, 0, 0, 0, 1, 1.0f, 0.0f, false, true);

    // ---- attention weights a_r / a_i ----
    k_logits<<<BS, 256, 0, stream>>>(r1, i1, r2, i2, logits);
    k_softmax<<<2 * B_, 256, 0, stream>>>(logits, 1.0f);

    // ---- softmax-weighted outer products: M = A^T diag(a) B as WMMA GEMMs ----
    k_scale4<<<(nSBE + TPB - 1) / TPB, TPB, 0, stream>>>(logits, r2, i2, wr2, wi2, ur2, ui2);
    launch_gemm(stream, wr2, r2, nullptr, MR, E_, E_, S_, E_, E_, E_, SEl, SEl, EE, B_,  1.0f, 0.0f, true, false);
    launch_gemm(stream, wi2, i2, nullptr, MR, E_, E_, S_, E_, E_, E_, SEl, SEl, EE, B_, -1.0f, 1.0f, true, false);
    launch_gemm(stream, ui2, r2, nullptr, MI, E_, E_, S_, E_, E_, E_, SEl, SEl, EE, B_,  1.0f, 0.0f, true, false);
    launch_gemm(stream, ur2, i2, nullptr, MI, E_, E_, S_, E_, E_, E_, SEl, SEl, EE, B_,  1.0f, 1.0f, true, false);

    // ---- rank-1 measurement, conv+pool, FCs ----
    k_projnorm<<<2, 256, 0, stream>>>(F(IN_PROJ), prn, pin);
    k_matvec<<<(B_ * E_ + TPB - 1) / TPB, TPB, 0, stream>>>(MR, MI, prn, pin, t1, t2, t3, t4);
    k_pv<<<(nBEE + TPB - 1) / TPB, TPB, 0, stream>>>(t1, t2, t3, t4, prn, pin, PvR, PvI);
    k_convpool<<<dim3(E_ - 2, B_, 2), 256, 0, stream>>>(PvR, PvI, F(IN_C1W), F(IN_C1B),
                                                        F(IN_C2W), F(IN_C2B), pooled);
    k_fc1<<<1, 256, 0, stream>>>(pooled, F(IN_FC1W), F(IN_FC1B), h10);
    k_fc2<<<1, 32, 0, stream>>>(h10, F(IN_FC2W), F(IN_FC2B), (float*)d_out);
}
